// MoELayer_76433237999752
// MI455X (gfx1250) — matrix operand order repaired and checked by reading, not compiled, not run on hardware
//
#include <hip/hip_runtime.h>
#include <stdint.h>

// ---------------- problem constants (match reference) ----------------
#define D_MODEL   1024
#define D_HIDDEN  4096
#define NUM_EXP   8
#define TOKENS    4096        // BATCH * SEQ = 2 * 2048
#define MT        16          // tokens per FFN workgroup (one 16-col N tile)
#define HC        128         // hidden-chunk width (8 waves * 16 rows)

typedef __attribute__((ext_vector_type(16))) __bf16 v16bf;
typedef __attribute__((ext_vector_type(8)))  float  v8f;

static __device__ __forceinline__ v8f wmma_bf16(v16bf a, v16bf b, v8f c) {
    return __builtin_amdgcn_wmma_f32_16x16x32_bf16(
        /*neg_a=*/false, a, /*neg_b=*/false, b,
        /*c_mod=*/(short)0, c, /*reuse_a=*/false, /*reuse_b=*/false);
}

static __device__ __forceinline__ void wait_async0() {
#if __has_builtin(__builtin_amdgcn_s_wait_asynccnt)
    __builtin_amdgcn_s_wait_asynccnt(0);
#else
    asm volatile("s_wait_asynccnt 0x0" ::: "memory");
#endif
}

// =====================================================================
// Kernel 0: tiled transpose + fp32 -> bf16 weight conversion.
//   src: [E][R][C] fp32   ->   dst: [E][C][R] bf16
// One-time prep (~17 us at HBM peak); makes every hot-loop fragment a
// contiguous 16-byte bf16 load (weights become the WMMA A operand).
// =====================================================================
__global__ __launch_bounds__(256) void transpose_bf16(
    const float* __restrict__ src, __bf16* __restrict__ dst, int R, int C)
{
    __shared__ float tile[32][33];
    const int e  = blockIdx.z;
    const float* s = src + (size_t)e * R * C;
    __bf16*      d = dst + (size_t)e * R * C;
    const int c0 = blockIdx.x * 32, r0 = blockIdx.y * 32;
    const int tx = threadIdx.x, ty = threadIdx.y;        // 32 x 8
#pragma unroll
    for (int i = 0; i < 32; i += 8)
        tile[ty + i][tx] = s[(size_t)(r0 + ty + i) * C + (c0 + tx)];
    __syncthreads();
#pragma unroll
    for (int i = 0; i < 32; i += 8)
        d[(size_t)(c0 + ty + i) * R + (r0 + tx)] = (__bf16)tile[tx][ty + i];
}

// =====================================================================
// Kernel 1: gating. One wave (32 lanes) per token.
//  - logits = x @ Wg + bg              -> logits slice of d_out
//  - top-2 + softmax over the two vals -> (expert, weight) pairs
//  - scatter token into per-expert lists (atomic counters in ws)
//  - convert token's x row to bf16 into ws (read once, reused by both experts)
// =====================================================================
__global__ __launch_bounds__(256) void moe_gate(
    const float* __restrict__ x, const float* __restrict__ Wg,
    const float* __restrict__ bg, float* __restrict__ logits_out,
    int* __restrict__ counts, int* __restrict__ tok_list,
    float* __restrict__ tok_w, __bf16* __restrict__ Xb)
{
    const int wave = threadIdx.x >> 5;
    const int lane = threadIdx.x & 31;
    const int t    = blockIdx.x * (blockDim.x >> 5) + wave;
    if (t >= TOKENS) return;

    const float* xr = x + (size_t)t * D_MODEL;

    float acc[NUM_EXP];
#pragma unroll
    for (int e = 0; e < NUM_EXP; ++e) acc[e] = 0.f;

    for (int d = lane; d < D_MODEL; d += 32) {
        float xv = xr[d];
        const float* wr = Wg + (size_t)d * NUM_EXP;
#pragma unroll
        for (int e = 0; e < NUM_EXP; ++e) acc[e] += xv * wr[e];
    }
#pragma unroll
    for (int e = 0; e < NUM_EXP; ++e) {
#pragma unroll
        for (int off = 16; off > 0; off >>= 1)
            acc[e] += __shfl_xor(acc[e], off, 32);
    }

    for (int d = lane; d < D_MODEL; d += 32)
        Xb[(size_t)t * D_MODEL + d] = (__bf16)xr[d];

    if (lane == 0) {
        float lg[NUM_EXP];
#pragma unroll
        for (int e = 0; e < NUM_EXP; ++e) {
            lg[e] = acc[e] + bg[e];
            logits_out[(size_t)t * NUM_EXP + e] = lg[e];
        }
        int e1 = 0;
#pragma unroll
        for (int e = 1; e < NUM_EXP; ++e) if (lg[e] > lg[e1]) e1 = e;
        int e2 = (e1 == 0) ? 1 : 0;
#pragma unroll
        for (int e = 0; e < NUM_EXP; ++e)
            if (e != e1 && lg[e] > lg[e2]) e2 = e;
        float p2 = __expf(lg[e2] - lg[e1]);
        float s  = 1.f + p2;
        float w1 = 1.f / s, w2 = p2 / s;

        int p = atomicAdd(&counts[e1], 1);
        tok_list[e1 * TOKENS + p] = t;  tok_w[e1 * TOKENS + p] = w1;
        p = atomicAdd(&counts[e2], 1);
        tok_list[e2 * TOKENS + p] = t;  tok_w[e2 * TOKENS + p] = w2;
    }
}

// =====================================================================
// Kernel 2: fused expert FFN on gathered tokens.
// grid = (TOKENS/MT, NUM_EXP), block = 256 (8 waves).
// Weights are the A operand (K contiguous -> 2x global b128 per frag),
// activations the B operand (lane = token, K contiguous in LDS ->
// 2x ds b128 per frag). hT/yT tiles: rows = hidden/dmodel, cols = tokens.
//   loop hidden chunks of 128:
//     stage1: hT(128 x 16) = relu(W1t[chunk,:] @ XsT + b1)  (1 acc/wave)
//     stage2: yT(1024 x16) += W2t[:,chunk] @ hT             (8 acc/wave)
//   out[t,:] += w_t * (y + b2)  via global fp32 atomics
// =====================================================================
__global__ __launch_bounds__(256) void moe_ffn(
    const __bf16* __restrict__ Xb,
    const __bf16* __restrict__ W1t,   // [E][H][D] bf16 (= W1^T per expert)
    const float*  __restrict__ b1,
    const __bf16* __restrict__ W2t,   // [E][D][H] bf16 (= W2^T per expert)
    const float*  __restrict__ b2,
    const int* __restrict__ counts, const int* __restrict__ tok_list,
    const float* __restrict__ tok_w, float* __restrict__ out)
{
    const int e     = blockIdx.y;
    const int Ne    = counts[e];
    const int slot0 = blockIdx.x * MT;
    if (slot0 >= Ne) return;                      // uniform early exit

    __shared__ __align__(16) __bf16 Xs[MT][D_MODEL];  // 32 KB, tokens x d
    __shared__ __align__(16) __bf16 Hs[MT][HC];       //  4 KB, tokens x hidden
    __shared__ int    toks[MT];
    __shared__ float  wts[MT];

    const int tid  = threadIdx.x;
    const int wave = tid >> 5;
    const int lane = tid & 31;
    const int l16  = lane & 15;       // A row sub-index / token column
    const int hi16 = lane >> 4;       // lane half select
    const int kb   = hi16 * 8;        // A-frag K sub-offset (halves)
    const int klo  = hi16 * 16;       // unused; B frags also use kb

    if (tid < MT) {
        int s = slot0 + tid;
        if (s < Ne) { toks[tid] = tok_list[e * TOKENS + s];
                      wts[tid]  = tok_w[e * TOKENS + s]; }
        else        { toks[tid] = -1; wts[tid] = 0.f; }
    }
    __syncthreads();

    // ---- gather 16 bf16 token rows into LDS via async LDS loads ----
    {
        int row = tid >> 4, sub = tid & 15;       // 16 threads/row, 16B chunks
        int tt  = toks[row];
        if (tt >= 0) {
            const __bf16* g = Xb + (size_t)tt * D_MODEL + sub * 8;
            unsigned lp = (unsigned)(uintptr_t)&Xs[row][sub * 8];
#pragma unroll
            for (int j = 0; j < 8; ++j) {
                const __bf16* gp = g + j * 128;   // +256 B per chunk
                unsigned      lo = lp + j * 256;
                asm volatile("global_load_async_to_lds_b128 %0, %1, off"
                             :: "v"(lo), "v"(gp) : "memory");
            }
            wait_async0();
        } else {
            uint4  z  = make_uint4(0, 0, 0, 0);
            uint4* dz = (uint4*)&Xs[row][sub * 8];
#pragma unroll
            for (int j = 0; j < 8; ++j) dz[j * 16] = z;
        }
    }
    __syncthreads();

    v8f acc2[8];
#pragma unroll
    for (int f = 0; f < 8; ++f) acc2[f] = {};

    const size_t eW   = (size_t)e * D_MODEL * D_HIDDEN;
    const __bf16* W1e = W1t + eW;                 // [H][D]
    const __bf16* W2e = W2t + eW;                 // [D][H]

    for (int hc = 0; hc < D_HIDDEN; hc += HC) {
        // ---------------- stage 1: hT chunk ----------------
        v8f acc1 = {};
        const int     hrow = hc + wave * 16 + l16;          // A row (hidden)
        const __bf16* arow = W1e + (size_t)hrow * D_MODEL;  // K contiguous
        __builtin_prefetch(arow + (size_t)HC * D_MODEL, 0, 1); // next chunk's row
        for (int k = 0; k < D_MODEL; k += 32) {
            union { uint4 u[2]; v16bf v; } A, B;
            A.u[0] = *(const uint4*)(arow + k + kb);
            A.u[1] = *(const uint4*)(arow + k + 16 + kb);
            B.u[0] = *(const uint4*)&Xs[l16][k + kb];
            B.u[1] = *(const uint4*)&Xs[l16][k + 16 + kb];
            acc1 = wmma_bf16(A.v, B.v, acc1);
        }
        __syncthreads();   // all waves done reading Hs from previous chunk

        // bias + relu; D tile rows = hidden, cols = tokens
#pragma unroll
        for (int r = 0; r < 8; ++r) {
            int   hloc = wave * 16 + r + 8 * hi16;           // hidden in chunk
            float hv   = acc1[r] + b1[e * D_HIDDEN + hc + hloc];
            hv = hv > 0.f ? hv : 0.f;                        // relu
            Hs[l16][hloc] = (__bf16)hv;                      // token-major
        }
        __syncthreads();

        // ---------------- stage 2: yT += W2t[:,chunk] @ hT ----------------
        for (int k2 = 0; k2 < HC; k2 += 32) {
            union { uint4 u[2]; v16bf v; } B2;
            B2.u[0] = *(const uint4*)&Hs[l16][k2 + kb];
            B2.u[1] = *(const uint4*)&Hs[l16][k2 + 16 + kb];
#pragma unroll
            for (int f = 0; f < 8; ++f) {
                const int     drow = f * 128 + wave * 16 + l16;  // A row (dmodel)
                const __bf16* a2   = W2e + (size_t)drow * D_HIDDEN + hc + k2;
                union { uint4 u[2]; v16bf v; } A2;
                A2.u[0] = *(const uint4*)(a2 + kb);
                A2.u[1] = *(const uint4*)(a2 + 16 + kb);
                acc2[f] = wmma_bf16(A2.v, B2.v, acc2[f]);
            }
        }
    }

    // ---------------- epilogue: out[t] += w * (y + b2) ----------------
    const int   tt = toks[l16];
    const float w  = wts[l16];
    if (tt >= 0) {
#pragma unroll
        for (int f = 0; f < 8; ++f) {
#pragma unroll
            for (int r = 0; r < 8; ++r) {
                int   d  = f * 128 + wave * 16 + r + 8 * hi16;
                float yv = acc2[f][r] + b2[e * D_MODEL + d];
                atomicAdd(&out[(size_t)tt * D_MODEL + d], w * yv);
            }
        }
    }
}

// =====================================================================
// host launcher
// =====================================================================
extern "C" void kernel_launch(void* const* d_in, const int* in_sizes, int n_in,
                              void* d_out, int out_size, void* d_ws, size_t ws_size,
                              hipStream_t stream) {
    (void)in_sizes; (void)n_in; (void)out_size; (void)ws_size;
    const float* x  = (const float*)d_in[0];
    const float* Wg = (const float*)d_in[1];
    const float* bg = (const float*)d_in[2];
    const float* W1 = (const float*)d_in[3];
    const float* b1 = (const float*)d_in[4];
    const float* W2 = (const float*)d_in[5];
    const float* b2 = (const float*)d_in[6];

    float* out    = (float*)d_out;                          // [T, D]
    float* logits = out + (size_t)TOKENS * D_MODEL;         // [T, E]

    // ---- workspace layout (~137 MB) ----
    const size_t W_ELEMS = (size_t)NUM_EXP * D_MODEL * D_HIDDEN; // 33.5M
    char*   ws       = (char*)d_ws;
    size_t  off      = 0;
    int*    counts   = (int*)(ws + off);   off += 256;
    int*    tok_list = (int*)(ws + off);   off += (size_t)NUM_EXP * TOKENS * 4;
    float*  tok_w    = (float*)(ws + off); off += (size_t)NUM_EXP * TOKENS * 4;
    __bf16* Xb       = (__bf16*)(ws + off); off += (size_t)TOKENS * D_MODEL * 2;
    __bf16* W1t      = (__bf16*)(ws + off); off += W_ELEMS * 2;   // [E][H][D]
    __bf16* W2t      = (__bf16*)(ws + off); off += W_ELEMS * 2;   // [E][D][H]

    hipMemsetAsync(counts, 0, 256, stream);
    hipMemsetAsync(out, 0, (size_t)TOKENS * D_MODEL * sizeof(float), stream);

    // weight prep: W1 [E][D][H] -> W1t [E][H][D]; W2 [E][H][D dims swapped]
    transpose_bf16<<<dim3(D_HIDDEN / 32, D_MODEL / 32, NUM_EXP),
                     dim3(32, 8), 0, stream>>>(W1, W1t, D_MODEL, D_HIDDEN);
    transpose_bf16<<<dim3(D_MODEL / 32, D_HIDDEN / 32, NUM_EXP),
                     dim3(32, 8), 0, stream>>>(W2, W2t, D_HIDDEN, D_MODEL);

    moe_gate<<<TOKENS / 8, 256, 0, stream>>>(x, Wg, bg, logits,
                                             counts, tok_list, tok_w, Xb);
    moe_ffn<<<dim3(TOKENS / MT, NUM_EXP), 256, 0, stream>>>(
        Xb, W1t, b1, W2t, b2, counts, tok_list, tok_w, out);
}
